// SelfAttentionBlock_47811575939253
// MI455X (gfx1250) — matrix operand order repaired and checked by reading, not compiled, not run on hardware
//
#include <hip/hip_runtime.h>
#include <stdint.h>

// ---------------------------------------------------------------------------
// SelfAttentionBlock for MI455X (gfx1250, wave32, WMMA 16x16x32 f16)
//   B=4, H=W=64 -> N=4096, C=256, CK=32
//   Stage 0: weights -> fp16 transposed (one-time, L2 resident)
//   Stage 1: projections; ONE wave computes all 20 col-tiles of a 16-row group
//            (A fragment loaded/converted once per K-chunk, 20 WMMAs)
//   Stage 2: fused flash-attention; per-block async LDS staging of K/V chunks
//            (global_load_async_to_lds_b128 + s_wait_asynccnt, double buffered)
// ---------------------------------------------------------------------------

typedef _Float16 v8h  __attribute__((ext_vector_type(8)));
typedef _Float16 v16h __attribute__((ext_vector_type(16)));
typedef float    v8f  __attribute__((ext_vector_type(8)));

#define BATCH 4
#define NPIX  4096
#define CHAN  256
#define CKEY  32

static __device__ __forceinline__ v16h cat8(v8h a, v8h b) {
  return __builtin_shufflevector(a, b, 0,1,2,3,4,5,6,7,8,9,10,11,12,13,14,15);
}

static __device__ __forceinline__ v8f wmma_f16(v16h a, v16h b, v8f c) {
  // D = A(16x32 f16) * B(32x16 f16) + C(16x16 f32)
  return __builtin_amdgcn_wmma_f32_16x16x32_f16(false, a, false, b, (short)0, c,
                                                false, false);
}

// Async global->LDS copy of 16 bytes (ASYNCcnt tracked).
// LDS byte offset == low 32 bits of the generic shared pointer (ISA aperture rule).
static __device__ __forceinline__ void async_b128(void* lds_dst, const void* gsrc) {
  uint32_t lds = (uint32_t)(uintptr_t)lds_dst;
  uint64_t ga  = (uint64_t)(uintptr_t)gsrc;
  asm volatile("global_load_async_to_lds_b128 %0, %1, off"
               :: "v"(lds), "v"(ga) : "memory");
}

static __device__ __forceinline__ void wait_async_all() {
  asm volatile("s_wait_asynccnt 0" ::: "memory");
}

// ---------------------------------------------------------------------------
// Stage 0: convert weights to fp16, transposed: WT[c_out][c_in]
//   WfT,WgT: [32][256]   WhT: [256][256]
// ---------------------------------------------------------------------------
__global__ void __launch_bounds__(256)
prep_weights(const float* __restrict__ Wf, const float* __restrict__ Wg,
             const float* __restrict__ Wh,
             _Float16* __restrict__ WfT, _Float16* __restrict__ WgT,
             _Float16* __restrict__ WhT)
{
  const int id = blockIdx.x * 256 + threadIdx.x;   // 0 .. 81919
  if (id < 32 * 256) {
    const int c = id >> 8, k = id & 255;
    WfT[id] = (_Float16)Wf[k * CKEY + c];
  } else if (id < 2 * 32 * 256) {
    const int j = id - 32 * 256;
    const int c = j >> 8, k = j & 255;
    WgT[j] = (_Float16)Wg[k * CKEY + c];
  } else {
    const int j = id - 2 * 32 * 256;
    const int c = j >> 8, k = j & 255;
    WhT[j] = (_Float16)Wh[k * CHAN + c];
  }
}

// ---------------------------------------------------------------------------
// Stage 1: projections. One wave = one 16-row group, ALL 20 col tiles:
//   t<16 -> hh col-tile t ; t=16,17 -> f ; t=18,19 -> g
// hh stored TRANSPOSED: hhT[b][c][n] ; f,g row-major fp16 [b][n][32]
// ---------------------------------------------------------------------------
__global__ void __launch_bounds__(256)
proj_kernel(const float* __restrict__ x,
            const _Float16* __restrict__ WfT, const float* __restrict__ bf,
            const _Float16* __restrict__ WgT, const float* __restrict__ bg,
            const _Float16* __restrict__ WhT, const float* __restrict__ bh,
            _Float16* __restrict__ fbuf, _Float16* __restrict__ gbuf,
            _Float16* __restrict__ hhT)
{
  const int lane   = threadIdx.x & 31;
  const int wave   = threadIdx.x >> 5;
  const int rowgrp = blockIdx.x * 8 + wave;     // 0 .. 1023
  const int b  = rowgrp >> 8;                   // 256 row-groups per batch
  const int n0 = (rowgrp & 255) << 4;
  const int hs = lane >> 4;
  const int lm = lane & 15;

  // 20 accumulators, bias-initialized (bias broadcast along rows)
  v8f acc[20];
  #pragma unroll
  for (int t = 0; t < 20; ++t) {
    const float* bias = (t < 16) ? bh : (t < 18 ? bf : bg);
    const int col0 = (t < 16) ? t * 16 : (t < 18 ? (t - 16) * 16 : (t - 18) * 16);
    const float bv = bias[col0 + lm];
    #pragma unroll
    for (int i = 0; i < 8; ++i) acc[t][i] = bv;
  }

  const float* xrow = x + (size_t)(b * NPIX + n0 + lm) * CHAN;

  for (int kc = 0; kc < 8; ++kc) {              // K = 256 in chunks of 32
    // A (16x32 f16), loaded & converted ONCE per chunk, reused by 20 WMMAs:
    // lanes 0-15 hold K {0..15}, lanes 16-31 {16..31}
    const int ka = kc * 32 + hs * 16;
    v16h A;
    #pragma unroll
    for (int i = 0; i < 8; ++i) {
      A[i]     = (_Float16)xrow[ka + i];
      A[i + 8] = (_Float16)xrow[ka + 8 + i];
    }
    // B (32x16 f16): lane -> out col, K {0..15}/{16..31} per half; contiguous
    #pragma unroll
    for (int t = 0; t < 20; ++t) {
      const _Float16* WT = (t < 16) ? WhT : (t < 18 ? WfT : WgT);
      const int col0 = (t < 16) ? t * 16 : (t < 18 ? (t - 16) * 16 : (t - 18) * 16);
      v16h Bm = *(const v16h*)(WT + (size_t)(col0 + lm) * CHAN + kc * 32 + hs * 16);
      acc[t] = wmma_f16(A, Bm, acc[t]);
    }
  }

  // D layout: VGPR v, lane -> col lm, row M = v + 8*hs
  #pragma unroll
  for (int t = 0; t < 16; ++t) {                // hh tiles, transposed b128 store
    _Float16* dst = hhT + (size_t)(b * CHAN + t * 16 + lm) * NPIX + n0 + 8 * hs;
    v8h pk;
    #pragma unroll
    for (int v = 0; v < 8; ++v) pk[v] = (_Float16)acc[t][v];
    *(v8h*)dst = pk;
  }
  #pragma unroll
  for (int t = 16; t < 20; ++t) {               // f,g tiles, row-major
    _Float16* dst = (t < 18) ? fbuf : gbuf;
    const int col0 = (t < 18) ? (t - 16) * 16 : (t - 18) * 16;
    #pragma unroll
    for (int v = 0; v < 8; ++v) {
      const int row = n0 + v + 8 * hs;
      dst[(size_t)(b * NPIX + row) * CKEY + col0 + lm] = (_Float16)acc[t][v];
    }
  }
}

// ---------------------------------------------------------------------------
// Stage 2: fused flash attention + residual.
// Block = 4 waves = 4 query tiles of the same batch. Per 32-key chunk the block
// async-stages hh (16KB) and f (2KB) slabs into double-buffered LDS; every
// wave's B fragments then come from LDS (ds_load_b128).
// ---------------------------------------------------------------------------
__global__ void __launch_bounds__(128)
attn_kernel(const float* __restrict__ x,
            const _Float16* __restrict__ fbuf,
            const _Float16* __restrict__ gbuf,
            const _Float16* __restrict__ hhT,
            float* __restrict__ out)
{
  __shared__ __align__(64) _Float16 hstage[2][CHAN * 32];  // 2 x 16 KB
  __shared__ __align__(64) _Float16 fstage[2][32 * 32];    // 2 x  2 KB
  __shared__ __align__(64) _Float16 pshare[4][16 * 32];    // 4 x  1 KB

  const int tid   = threadIdx.x;
  const int lane  = tid & 31;
  const int wave  = tid >> 5;
  const int qtile = blockIdx.x * 4 + wave;      // 0 .. 1023
  const int b     = (blockIdx.x * 4) >> 8;      // uniform: same batch per block
  const int q0    = (qtile & 255) << 4;
  const int hs    = lane >> 4;
  const int lm    = lane & 15;
  _Float16* ps = pshare[wave];

  const _Float16* hbase = hhT  + (size_t)b * CHAN * NPIX;
  const _Float16* fbase = fbuf + (size_t)b * NPIX * CKEY;

  // ---- async stage of one 32-key chunk: hh slab [256c][32k], f slab [32j][32k]
  auto stage_chunk = [&](int jb, int sel) {
    _Float16* hd = hstage[sel];
    const _Float16* hsrc = hbase + jb;
    #pragma unroll
    for (int i = 0; i < 8; ++i) {
      const int xid = tid + 128 * i;            // 1024 x 16B transfers
      const int c = xid >> 2, q = xid & 3;
      async_b128(hd + c * 32 + q * 8, hsrc + (size_t)c * NPIX + q * 8);
    }
    _Float16* fd = fstage[sel];
    const _Float16* fsrc = fbase + (size_t)jb * CKEY;
    const int j = tid >> 2, q = tid & 3;        // 128 x 16B transfers
    async_b128(fd + j * 32 + q * 8, fsrc + j * CKEY + q * 8);
  };

  // A fragment of g (queries), loaded once
  const _Float16* grow = gbuf + (size_t)(b * NPIX + q0 + lm) * CKEY;
  const int ka = hs * 16;
  v16h ga = cat8(*(const v8h*)(grow + ka), *(const v8h*)(grow + ka + 8));

  v8f acc[16];
  #pragma unroll
  for (int t = 0; t < 16; ++t)
    #pragma unroll
    for (int v = 0; v < 8; ++v) acc[t][v] = 0.0f;

  float m[8], lsum[8];
  #pragma unroll
  for (int v = 0; v < 8; ++v) { m[v] = -3.0e38f; lsum[v] = 0.0f; }

  int cur = 0;
  stage_chunk(0, 0);
  wait_async_all();
  __syncthreads();

  for (int jb = 0; jb < NPIX; jb += 32) {
    const int nxt = cur ^ 1;
    if (jb + 32 < NPIX) stage_chunk(jb + 32, nxt);   // overlap with compute

    // ---- s = g . f^T from LDS f slab: B[k][j] = f[jb+j][k]
    const _Float16* fs = fstage[cur];
    v16h fb0 = *(const v16h*)(fs + lm * 32 + hs * 16);
    v16h fb1 = *(const v16h*)(fs + (16 + lm) * 32 + hs * 16);
    v8f z = {0.f, 0.f, 0.f, 0.f, 0.f, 0.f, 0.f, 0.f};
    v8f s0 = wmma_f16(ga, fb0, z);
    v8f s1 = wmma_f16(ga, fb1, z);

    // ---- online softmax (row stats live in the 16-lane groups of D layout)
    float scale[8];
    #pragma unroll
    for (int v = 0; v < 8; ++v) {
      float rmax = fmaxf(s0[v], s1[v]);
      #pragma unroll
      for (int msk = 8; msk >= 1; msk >>= 1)
        rmax = fmaxf(rmax, __shfl_xor(rmax, msk, 32));
      const float mn = fmaxf(m[v], rmax);
      const float sc = __expf(m[v] - mn);
      m[v] = mn;
      const float p0 = __expf(s0[v] - mn);
      const float p1 = __expf(s1[v] - mn);
      s0[v] = p0; s1[v] = p1;
      float rs = p0 + p1;
      #pragma unroll
      for (int msk = 8; msk >= 1; msk >>= 1)
        rs += __shfl_xor(rs, msk, 32);
      lsum[v] = lsum[v] * sc + rs;
      scale[v] = sc;
    }
    #pragma unroll
    for (int t = 0; t < 16; ++t)
      #pragma unroll
      for (int v = 0; v < 8; ++v) acc[t][v] *= scale[v];

    // ---- restage P: D layout -> LDS [row][col] -> A layout
    #pragma unroll
    for (int v = 0; v < 8; ++v) {
      ps[(v + 8 * hs) * 32 + lm]      = (_Float16)s0[v];
      ps[(v + 8 * hs) * 32 + 16 + lm] = (_Float16)s1[v];
    }
    asm volatile("s_wait_dscnt 0" ::: "memory");
    const _Float16* pr = ps + lm * 32 + hs * 16;
    v16h pa = cat8(*(const v8h*)pr, *(const v8h*)(pr + 8));

    // ---- o += P * hh from LDS hh slab (B contiguous per lane)
    const _Float16* hsl = hstage[cur];
    #pragma unroll
    for (int t = 0; t < 16; ++t) {
      v16h hv = *(const v16h*)(hsl + (t * 16 + lm) * 32 + hs * 16);
      acc[t] = wmma_f16(pa, hv, acc[t]);
    }

    wait_async_all();      // this wave's stage of `nxt` has landed
    __syncthreads();       // all waves staged `nxt` and finished reading `cur`
    cur = nxt;
  }

  // ---- epilogue: normalize, residual add
  float inv[8];
  #pragma unroll
  for (int v = 0; v < 8; ++v) inv[v] = 1.0f / lsum[v];

  #pragma unroll
  for (int t = 0; t < 16; ++t) {
    #pragma unroll
    for (int v = 0; v < 8; ++v) {
      const int row = q0 + v + 8 * hs;
      const size_t idx = (size_t)(b * NPIX + row) * CHAN + t * 16 + lm;
      out[idx] = x[idx] + acc[t][v] * inv[v];
    }
  }
}

// ---------------------------------------------------------------------------
extern "C" void kernel_launch(void* const* d_in, const int* in_sizes, int n_in,
                              void* d_out, int out_size, void* d_ws, size_t ws_size,
                              hipStream_t stream) {
  const float* x  = (const float*)d_in[0];
  const float* Wf = (const float*)d_in[1];
  const float* bf = (const float*)d_in[2];
  const float* Wg = (const float*)d_in[3];
  const float* bg = (const float*)d_in[4];
  const float* Wh = (const float*)d_in[5];
  const float* bh = (const float*)d_in[6];
  float* out = (float*)d_out;

  char* ws = (char*)d_ws;
  _Float16* fbuf = (_Float16*)(ws);                        // 1 MB [4][4096][32]
  _Float16* gbuf = (_Float16*)(ws + (size_t)(1u << 20));   // 1 MB [4][4096][32]
  _Float16* hhT  = (_Float16*)(ws + (size_t)(2u << 20));   // 8 MB [4][256][4096]
  _Float16* WfT  = (_Float16*)(ws + (size_t)(10u << 20));            // 16 KB
  _Float16* WgT  = (_Float16*)(ws + (size_t)(10u << 20) + (16u<<10));// 16 KB
  _Float16* WhT  = (_Float16*)(ws + (size_t)(10u << 20) + (32u<<10));// 128 KB

  prep_weights<<<320, 256, 0, stream>>>(Wf, Wg, Wh, WfT, WgT, WhT);
  // 1024 row-groups, one wave each, 8 waves/block
  proj_kernel<<<128, 256, 0, stream>>>(x, WfT, bf, WgT, bg, WhT, bh,
                                       fbuf, gbuf, hhT);
  // 1024 query tiles, 4 waves/block
  attn_kernel<<<256, 128, 0, stream>>>(x, fbuf, gbuf, hhT, out);
}
